// NeuralCE_Ising_28149215658676
// MI455X (gfx1250) — compile-verified, matrix operand order checked by reading, not run on hardware
//
#include <hip/hip_runtime.h>
#include <hip/hip_bf16.h>

// ---------------- WMMA types / helpers (gfx1250, wave32) ----------------
typedef __attribute__((ext_vector_type(16))) _Float16 v16h;
typedef __attribute__((ext_vector_type(8)))  _Float16 v8h;
typedef __attribute__((ext_vector_type(8)))  float    v8f;

#define DEV __device__ __forceinline__

// Load a 16x32 f16 A/B fragment from row-major LDS.
// ISA layout: lanes 0-15 hold row=lane, K base 0; lanes 16-31 hold row=lane-16,
// K base 8. VGPR0-3 = K[base..base+7], VGPR4-7 = K[base+16..base+23].
// Caller passes p = &buf[row*stride + ktile*32 + (lane>=16 ? 8 : 0)].
DEV v16h frag_ld(const _Float16* p) {
  v8h lo = *(const v8h*)(p);
  v8h hi = *(const v8h*)(p + 16);
  v16h a;
#pragma unroll
  for (int i = 0; i < 8; ++i) { a[i] = lo[i]; a[i + 8] = hi[i]; }
  return a;
}

DEV v8f wmma16(v16h a, v16h b, v8f c) {
  return __builtin_amdgcn_wmma_f32_16x16x32_f16(false, a, false, b, (short)0, c,
                                                false, false);
}

DEV float softplus_f(float x) { return (x > 20.f) ? x : log1pf(__expf(x)); }
DEV float sigmoid_f(float x)  { return 1.f / (1.f + __expf(-x)); }

// ---------------- LayerNorm (wave32: 2 elems/lane, butterfly reduce) -----
__global__ void __launch_bounds__(256) ln_kernel(const float* __restrict__ h,
                                                 const float* __restrict__ s,
                                                 const float* __restrict__ b,
                                                 _Float16* __restrict__ xout,
                                                 int n) {
  const int lane = threadIdx.x & 31;
  const int wave = threadIdx.x >> 5;
  const int row  = blockIdx.x * 8 + wave;
  if (row >= n) return;
  const float* hr = h + (size_t)row * 64;
  float a0 = hr[lane], a1 = hr[lane + 32];
  float sum = a0 + a1;
#pragma unroll
  for (int off = 16; off >= 1; off >>= 1) sum += __shfl_xor(sum, off, 32);
  float mu = sum * (1.f / 64.f);
  float d0 = a0 - mu, d1 = a1 - mu;
  float vs = d0 * d0 + d1 * d1;
#pragma unroll
  for (int off = 16; off >= 1; off >>= 1) vs += __shfl_xor(vs, off, 32);
  float rs = rsqrtf(vs * (1.f / 64.f) + 1e-6f);
  xout[(size_t)row * 64 + lane]      = (_Float16)(d0 * rs * s[lane] + b[lane]);
  xout[(size_t)row * 64 + lane + 32] = (_Float16)(d1 * rs * s[lane + 32] + b[lane + 32]);
}

// ---------------- Generic row-GEMM: out = act(A[n,ka] @ W[ka,NOUT] + bias) --
template <typename AT, int KPAD, int NOUT, int ACT>
__global__ void __launch_bounds__(256) wmma_gemm(
    const AT* __restrict__ A, int ka, const float* __restrict__ W,
    const float* __restrict__ bias, _Float16* __restrict__ outh,
    float* __restrict__ outf, int nrows) {
  __shared__ _Float16 sA[128 * KPAD];
  __shared__ _Float16 sW[NOUT * KPAD];  // transposed: [n][k]
  __shared__ float    sB[NOUT];
  const int tid  = threadIdx.x;
  const int row0 = blockIdx.x * 128;

  for (int t = tid; t < NOUT * KPAD; t += 256) {
    int n = t / KPAD, k = t - n * KPAD;
    sW[t] = (k < ka) ? (_Float16)W[(size_t)k * NOUT + n] : (_Float16)0.f;
  }
  for (int t = tid; t < NOUT; t += 256) sB[t] = bias[t];
  for (int t = tid; t < 128 * KPAD; t += 256) {
    int r = t / KPAD, k = t - r * KPAD;
    int gr = row0 + r;
    float v = (gr < nrows && k < ka) ? (float)A[(size_t)gr * ka + k] : 0.f;
    sA[t] = (_Float16)v;
  }
  __syncthreads();

  const int lane = tid & 31;
  const int wave = tid >> 5;
  const int n16  = lane & 15;
  const int hsel = lane >> 4;
  const int rowoff = hsel * 8;
  constexpr int KT = KPAD / 32;

  v16h af[KT];
#pragma unroll
  for (int kt = 0; kt < KT; ++kt)
    af[kt] = frag_ld(&sA[(wave * 16 + n16) * KPAD + kt * 32 + hsel * 8]);

#pragma unroll
  for (int nt = 0; nt < NOUT / 16; ++nt) {
    const int col = nt * 16 + n16;
    float bv = sB[col];
    v8f acc;
#pragma unroll
    for (int i = 0; i < 8; ++i) acc[i] = bv;
#pragma unroll
    for (int kt = 0; kt < KT; ++kt)
      acc = wmma16(af[kt], frag_ld(&sW[col * KPAD + kt * 32 + hsel * 8]), acc);
#pragma unroll
    for (int r = 0; r < 8; ++r) {
      int gr = row0 + wave * 16 + r + rowoff;
      if (gr < nrows) {
        float v = acc[r];
        if (ACT == 1) v = softplus_f(v);
        if (outh) outh[(size_t)gr * NOUT + col] = (_Float16)v;
        if (outf) outf[(size_t)gr * NOUT + col] = v;
      }
    }
  }
}

// ---------------- Fused CGCNN conv layer -------------------------------
// Block owns 16 atoms -> 384 edges -> 24 WMMA M-tiles (3 per wave).
// Fuses: phi_c GEMM (wave 0), edge GEMMs phi_n/phi_e, triple product,
// gate/mag GEMMs, neighbor reduce (ds_add_f32), h update, and next-layer
// LayerNorm in the epilogue. x-row gather uses CDNA5 async global->LDS.
__global__ void __launch_bounds__(256) conv_kernel(
    const _Float16* __restrict__ xin, const float* __restrict__ nbr_fea,
    const int* __restrict__ nbr_idx,
    const float* __restrict__ Wc, const float* __restrict__ bc,
    const float* __restrict__ Wn, const float* __restrict__ bn,
    const float* __restrict__ We, const float* __restrict__ be,
    const float* __restrict__ Wg, const float* __restrict__ bg,
    const float* __restrict__ Wm, const float* __restrict__ bm,
    float* __restrict__ h32, _Float16* __restrict__ h16,
    const float* __restrict__ lns_next, const float* __restrict__ lnb_next,
    _Float16* __restrict__ xnext, int n) {
  __shared__ _Float16 sWc[64 * 64], sWn[64 * 64], sWe[64 * 64], sWg[64 * 64],
      sWm[64 * 64];                   // transposed [n][k], f16
  __shared__ _Float16 sXn[384 * 64];  // gathered neighbor x rows (f16)
  __shared__ _Float16 sFe[384 * 64];  // nbr_fea padded; reused as inter
  __shared__ _Float16 sXc[16 * 64];   // own-atom x rows (for phi_c)
  __shared__ float sPhiC[16 * 64];
  __shared__ float sHacc[16 * 64];
  __shared__ float sBc[64], sBn[64], sBe[64], sBg[64], sBm[64];

  const int tid   = threadIdx.x;
  const int atom0 = blockIdx.x * 16;

  // weights -> LDS, f16, transposed [n][k] for contiguous B-fragments
  for (int t = tid; t < 4096; t += 256) {
    int k = t >> 6, nn = t & 63;
    int tr = nn * 64 + k;
    sWc[tr] = (_Float16)Wc[t];
    sWn[tr] = (_Float16)Wn[t];
    sWg[tr] = (_Float16)Wg[t];
    sWm[tr] = (_Float16)Wm[t];
    sWe[tr] = (k < 41) ? (_Float16)We[k * 64 + nn] : (_Float16)0.f;
  }
  for (int t = tid; t < 64; t += 256) {
    sBc[t] = bc[t]; sBn[t] = bn[t]; sBe[t] = be[t];
    sBg[t] = bg[t]; sBm[t] = bm[t];
  }
  for (int t = tid; t < 1024; t += 256) {
    int al = t >> 6;
    sXc[t]   = (atom0 + al < n) ? xin[(size_t)atom0 * 64 + t] : (_Float16)0.f;
    sHacc[t] = 0.f;
  }
  // zero-fill the K padding region of sFe (cols 41..63 of each edge row)
  for (int t = tid; t < 384 * 23; t += 256) {
    int e = t / 23, k = 41 + (t - e * 23);
    sFe[e * 64 + k] = (_Float16)0.f;
  }

  // nbr_fea: the block's 384 edge rows are one contiguous slab of
  // 384*41 = 15744 floats -> coalesced float4 loads, scatter-convert to f16.
  if (atom0 + 16 <= n) {
    const float4* slab = (const float4*)(nbr_fea + (size_t)atom0 * 24 * 41);
    for (int t = tid; t < 3936; t += 256) {
      float4 v = slab[t];
      int base = t * 4;
#pragma unroll
      for (int j = 0; j < 4; ++j) {
        int idx = base + j;
        int e = idx / 41;
        int k = idx - e * 41;
        sFe[e * 64 + k] = (_Float16)((const float*)&v)[j];
      }
    }
  } else {  // tail block: per-edge guarded scalar path
    for (int e = tid; e < 384; e += 256) {
      int al = e / 24, m = e - al * 24;
      int ga = atom0 + al;
      _Float16* fd = sFe + e * 64;
      if (ga < n) {
        const float* fs = nbr_fea + ((size_t)ga * 24 + m) * 41;
#pragma unroll
        for (int k = 0; k < 41; ++k) fd[k] = (_Float16)fs[k];
      } else {
#pragma unroll
        for (int k = 0; k < 41; ++k) fd[k] = (_Float16)0.f;
      }
    }
  }

  // gather neighbor x rows: CDNA5 async global->LDS (ASYNCcnt-tracked),
  // 8 x b128 per 128B row, bypassing the VGPR round-trip.
  for (int e = tid; e < 384; e += 256) {
    int al = e / 24, m = e - al * 24;
    int ga = atom0 + al;
    if (ga < n) {
      int nb = nbr_idx[(size_t)ga * 24 + m];
      unsigned lds0 = (unsigned)(uintptr_t)(sXn + (size_t)e * 64);
      unsigned long long g0 =
          (unsigned long long)(uintptr_t)(xin + (size_t)nb * 64);
#pragma unroll
      for (int q = 0; q < 8; ++q) {
        asm volatile("global_load_async_to_lds_b128 %0, %1, off"
                     :: "v"(lds0 + q * 16), "v"(g0 + q * 16)
                     : "memory");
      }
    } else {
      v8h z = {};
      v8h* dst = (v8h*)(sXn + (size_t)e * 64);
#pragma unroll
      for (int q = 0; q < 8; ++q) dst[q] = z;
    }
  }
  asm volatile("s_wait_asynccnt 0x0" ::: "memory");
  __syncthreads();

  const int lane   = tid & 31;
  const int wave   = tid >> 5;
  const int n16    = lane & 15;
  const int hsel   = lane >> 4;
  const int rowoff = hsel * 8;

  // fused phi_c = x @ Wc + bc for this block's 16 atoms (wave 0 only;
  // branch is wave-uniform so EXEC stays all-ones for the WMMAs)
  if (wave == 0) {
    const int arow = n16 * 64 + hsel * 8;
    v16h ca0 = frag_ld(&sXc[arow]);
    v16h ca1 = frag_ld(&sXc[arow + 32]);
#pragma unroll
    for (int nt = 0; nt < 4; ++nt) {
      const int col  = nt * 16 + n16;
      const int brow = col * 64 + hsel * 8;
      float bv = sBc[col];
      v8f acc;
#pragma unroll
      for (int i = 0; i < 8; ++i) acc[i] = bv;
      acc = wmma16(ca0, frag_ld(&sWc[brow]), acc);
      acc = wmma16(ca1, frag_ld(&sWc[brow + 32]), acc);
#pragma unroll
      for (int r = 0; r < 8; ++r)
        sPhiC[(r + rowoff) * 64 + col] = acc[r];
    }
  }
  __syncthreads();

  for (int tt = 0; tt < 3; ++tt) {
    const int mt   = wave * 3 + tt;      // M-tile 0..23 (wave-exclusive)
    const int row0 = mt * 16;
    const int arow = (row0 + n16) * 64 + hsel * 8;

    v16h xa0 = frag_ld(&sXn[arow]);
    v16h xa1 = frag_ld(&sXn[arow + 32]);
    v16h fa0 = frag_ld(&sFe[arow]);
    v16h fa1 = frag_ld(&sFe[arow + 32]);

    // phase 1: inter = phi_c * phi_n * phi_e (written over sFe rows)
#pragma unroll
    for (int nt = 0; nt < 4; ++nt) {
      const int col  = nt * 16 + n16;
      const int brow = col * 64 + hsel * 8;
      v8f an, ae;
      float bnv = sBn[col], bev = sBe[col];
#pragma unroll
      for (int i = 0; i < 8; ++i) { an[i] = bnv; ae[i] = bev; }
      an = wmma16(xa0, frag_ld(&sWn[brow]), an);
      an = wmma16(xa1, frag_ld(&sWn[brow + 32]), an);
      ae = wmma16(fa0, frag_ld(&sWe[brow]), ae);
      ae = wmma16(fa1, frag_ld(&sWe[brow + 32]), ae);
#pragma unroll
      for (int r = 0; r < 8; ++r) {
        int e  = row0 + r + rowoff;
        int al = e / 24;
        float pc = sPhiC[al * 64 + col];
        sFe[e * 64 + col] = (_Float16)(pc * an[r] * ae[r]);
      }
    }

    // phase 2: gate/mag GEMMs on inter, fused reduce over neighbor axis
    v16h ia0 = frag_ld(&sFe[arow]);
    v16h ia1 = frag_ld(&sFe[arow + 32]);
#pragma unroll
    for (int nt = 0; nt < 4; ++nt) {
      const int col  = nt * 16 + n16;
      const int brow = col * 64 + hsel * 8;
      v8f g, mm;
      float bgv = sBg[col], bmv = sBm[col];
#pragma unroll
      for (int i = 0; i < 8; ++i) { g[i] = bgv; mm[i] = bmv; }
      g  = wmma16(ia0, frag_ld(&sWg[brow]), g);
      g  = wmma16(ia1, frag_ld(&sWg[brow + 32]), g);
      mm = wmma16(ia0, frag_ld(&sWm[brow]), mm);
      mm = wmma16(ia1, frag_ld(&sWm[brow + 32]), mm);
#pragma unroll
      for (int r = 0; r < 8; ++r) {
        int e  = row0 + r + rowoff;
        int al = e / 24;
        atomicAdd(&sHacc[al * 64 + col], sigmoid_f(g[r]) * softplus_f(mm[r]));
      }
    }
  }
  __syncthreads();

  // epilogue: h += sum; optionally fused LayerNorm for the NEXT layer
  // (each wave owns 2 atom rows; wave32 butterfly reductions)
#pragma unroll
  for (int rr = 0; rr < 2; ++rr) {
    int al = wave * 2 + rr;
    int ga = atom0 + al;
    if (ga < n) {
      size_t base = (size_t)ga * 64;
      float v0 = h32[base + lane]      + sHacc[al * 64 + lane];
      float v1 = h32[base + lane + 32] + sHacc[al * 64 + lane + 32];
      h32[base + lane]      = v0;
      h32[base + lane + 32] = v1;
      h16[base + lane]      = (_Float16)v0;
      h16[base + lane + 32] = (_Float16)v1;
      if (lns_next) {
        float sum = v0 + v1;
#pragma unroll
        for (int off = 16; off >= 1; off >>= 1) sum += __shfl_xor(sum, off, 32);
        float mu = sum * (1.f / 64.f);
        float d0 = v0 - mu, d1 = v1 - mu;
        float vs = d0 * d0 + d1 * d1;
#pragma unroll
        for (int off = 16; off >= 1; off >>= 1) vs += __shfl_xor(vs, off, 32);
        float rs = rsqrtf(vs * (1.f / 64.f) + 1e-6f);
        xnext[base + lane] =
            (_Float16)(d0 * rs * lns_next[lane] + lnb_next[lane]);
        xnext[base + lane + 32] =
            (_Float16)(d1 * rs * lns_next[lane + 32] + lnb_next[lane + 32]);
      }
    }
  }
}

// ---------------- final dot(64->1) + mean pool -------------------------
__global__ void __launch_bounds__(256) readout_final(
    const _Float16* __restrict__ r2, const float* __restrict__ Wr3,
    const float* __restrict__ br3, float* __restrict__ out, int n) {
  __shared__ float sW[64];
  __shared__ float red[8];
  const int tid = threadIdx.x;
  if (tid < 64) sW[tid] = Wr3[tid];
  __syncthreads();
  int row = blockIdx.x * 256 + tid;
  float e = 0.f;
  if (row < n) {
    float s = br3[0];
    const _Float16* rr = r2 + (size_t)row * 64;
#pragma unroll
    for (int k = 0; k < 64; ++k) s += (float)rr[k] * sW[k];
    e = s;
  }
#pragma unroll
  for (int off = 16; off >= 1; off >>= 1) e += __shfl_xor(e, off, 32);
  const int lane = tid & 31, wave = tid >> 5;
  if (lane == 0) red[wave] = e;
  __syncthreads();
  if (tid == 0) {
    float t = 0.f;
#pragma unroll
    for (int w = 0; w < 8; ++w) t += red[w];
    atomicAdd(out, t * (1.0f / (float)n));
  }
}

// ---------------- launch ----------------------------------------------
extern "C" void kernel_launch(void* const* d_in, const int* in_sizes, int n_in,
                              void* d_out, int out_size, void* d_ws,
                              size_t ws_size, hipStream_t stream) {
  (void)n_in; (void)out_size; (void)ws_size;
  const float* atom_fea = (const float*)d_in[0];
  const float* nbr_fea  = (const float*)d_in[1];
  const int*   nbr_idx  = (const int*)d_in[2];
  const float* W_embed  = (const float*)d_in[3];
  const float* b_embed  = (const float*)d_in[4];
  const float* ln_s     = (const float*)d_in[5];
  const float* ln_b     = (const float*)d_in[6];
  const float* Wc = (const float*)d_in[7];  const float* bc = (const float*)d_in[8];
  const float* Wn = (const float*)d_in[9];  const float* bn = (const float*)d_in[10];
  const float* We = (const float*)d_in[11]; const float* be = (const float*)d_in[12];
  const float* Wg = (const float*)d_in[13]; const float* bg = (const float*)d_in[14];
  const float* Wm = (const float*)d_in[15]; const float* bm = (const float*)d_in[16];
  const float* Wr1 = (const float*)d_in[17]; const float* br1 = (const float*)d_in[18];
  const float* Wr2 = (const float*)d_in[19]; const float* br2 = (const float*)d_in[20];
  const float* Wr3 = (const float*)d_in[21]; const float* br3 = (const float*)d_in[22];

  const int N = in_sizes[0] / 92;  // 50000

  char* p = (char*)d_ws;
  auto carve = [&](size_t bytes) {
    char* r = p;
    p += (bytes + 255) & ~(size_t)255;
    return r;
  };
  float*    h32 = (float*)carve((size_t)N * 64 * sizeof(float));
  _Float16* h16 = (_Float16*)carve((size_t)N * 64 * sizeof(_Float16));
  _Float16* xA  = (_Float16*)carve((size_t)N * 64 * sizeof(_Float16));
  _Float16* xB  = (_Float16*)carve((size_t)N * 64 * sizeof(_Float16));
  _Float16* r1  = (_Float16*)carve((size_t)N * 128 * sizeof(_Float16));
  _Float16* r2  = (_Float16*)carve((size_t)N * 64 * sizeof(_Float16));

  const dim3 blk(256);
  const int gblocks = (N + 127) / 128;
  const int cblocks = (N + 15) / 16;

  // embed: h = atom_fea(92) @ W_embed + b  (K padded 92->96)
  wmma_gemm<float, 96, 64, 0><<<gblocks, blk, 0, stream>>>(
      atom_fea, 92, W_embed, b_embed, h16, h32, N);
  // first LayerNorm (subsequent ones are fused into conv epilogues)
  ln_kernel<<<(N + 7) / 8, blk, 0, stream>>>(h32, ln_s, ln_b, xA, N);

  const _Float16* xi[3] = {xA, xB, xA};
  _Float16*       xo[3] = {xB, xA, nullptr};
  for (int i = 0; i < 3; ++i) {
    const float* lns_nx = (i < 2) ? (ln_s + (i + 1) * 64) : nullptr;
    const float* lnb_nx = (i < 2) ? (ln_b + (i + 1) * 64) : nullptr;
    conv_kernel<<<cblocks, blk, 0, stream>>>(
        xi[i], nbr_fea, nbr_idx, Wc + (size_t)i * 4096, bc + i * 64,
        Wn + (size_t)i * 4096, bn + i * 64, We + (size_t)i * 41 * 64,
        be + i * 64, Wg + (size_t)i * 4096, bg + i * 64,
        Wm + (size_t)i * 4096, bm + i * 64, h32, h16, lns_nx, lnb_nx, xo[i], N);
  }

  // readout MLP with fused softplus
  wmma_gemm<_Float16, 64, 128, 1><<<gblocks, blk, 0, stream>>>(
      h16, 64, Wr1, br1, r1, nullptr, N);
  wmma_gemm<_Float16, 128, 64, 1><<<gblocks, blk, 0, stream>>>(
      r1, 128, Wr2, br2, r2, nullptr, N);

  hipMemsetAsync(d_out, 0, sizeof(float), stream);
  readout_final<<<(N + 255) / 256, blk, 0, stream>>>(r2, Wr3, br3,
                                                     (float*)d_out, N);
}
